// STGCNAttentionModel_56487409877028
// MI455X (gfx1250) — compile-verified, matrix operand order checked by reading, not run on hardware
//
#include <hip/hip_runtime.h>
#include <hip/hip_bf16.h>

// ---------------------------------------------------------------------------
// STGCN attention model for MI455X (gfx1250, wave32).
// Dense GEMMs -> v_wmma_f32_16x16x32_f16 with fragment-major f16 weights and
// b128 LDS fragment loads; graph segment-sum -> L2-resident f32 atomics.
// ---------------------------------------------------------------------------

typedef __attribute__((ext_vector_type(16))) _Float16 v16h;
typedef __attribute__((ext_vector_type(8)))  _Float16 v8h;
typedef __attribute__((ext_vector_type(8)))  float    v8f;

#define T_LEN   12
#define HID     64
#define CIN_D   16
#define NLAY    3
#define KCONV   192   // HID * 3 taps
#define ROWF    768   // HID * T_LEN  (per-node feature row, f32)
#define BN_EPS  1e-5f

// ------------------------------ helpers ------------------------------------

__device__ inline float wred_sum(float v) {
#pragma unroll
  for (int o = 16; o > 0; o >>= 1) v += __shfl_xor(v, o, 32);
  return v;
}
__device__ inline float wred_max(float v) {
#pragma unroll
  for (int o = 16; o > 0; o >>= 1) v = fmaxf(v, __shfl_xor(v, o, 32));
  return v;
}
__device__ inline void atomic_add_f32(float* p, float v) {
  __hip_atomic_fetch_add(p, v, __ATOMIC_RELAXED, __HIP_MEMORY_SCOPE_AGENT);
}
__device__ inline v16h cat16(v8h lo, v8h hi) {
  return __builtin_shufflevector(lo, hi, 0, 1, 2, 3, 4, 5, 6, 7,
                                 8, 9, 10, 11, 12, 13, 14, 15);
}
// K index inside a 32-wide chunk owned by (lane-half, fragment slot u, bit)
__device__ inline int frag_k(int half, int u, int bit) {
  return ((u >> 2) << 4) + (half << 3) + ((u & 3) << 1) + bit;
}

// ------------------------- small utility kernels ---------------------------

__global__ void k_fill(float* p, int n, float v) {
  int i = blockIdx.x * blockDim.x + threadIdx.x;
  int s = gridDim.x * blockDim.x;
  for (; i < n; i += s) p[i] = v;
}

// Convert weights to f16 in FRAGMENT-MAJOR order and fold BN into scale/bias.
// Layout: W*[l][chunk][tid][j]  (tid 0..127, j 0..15) -> one b128-pair per
// thread per chunk in the GEMM kernels.
__global__ void k_prep(const float* __restrict__ gcnW, const float* __restrict__ convW,
                       const float* __restrict__ bn_g, const float* __restrict__ bn_b,
                       const float* __restrict__ bn_m, const float* __restrict__ bn_v,
                       const float* __restrict__ conv_b,
                       _Float16* __restrict__ Wg, _Float16* __restrict__ Wc,
                       float* __restrict__ scl, float* __restrict__ sft) {
  int tid0 = blockIdx.x * blockDim.x + threadIdx.x;
  int str = gridDim.x * blockDim.x;
  // GCN weight (L,H,H)=[l][c][d]; GEMM B[k=c][d]; 2 chunks of K=32.
  for (int i = tid0; i < NLAY * 2 * 128 * 16; i += str) {
    int l = i / 4096, r = i & 4095;
    int ch = r >> 11, t = (r >> 4) & 127, j = r & 15;
    int wv = t >> 5, lane = t & 31, half = lane >> 4;
    int ncol = wv * 16 + (lane & 15);
    int k = ch * 32 + frag_k(half, j >> 1, j & 1);
    Wg[i] = (_Float16)gcnW[(l * HID + k) * HID + ncol];
  }
  // conv weight (L,H,H,3)=[l][d][c][tap]; GEMM B[k=tap*64+c][d]; 6 chunks.
  for (int i = tid0; i < NLAY * 6 * 128 * 16; i += str) {
    int l = i / 12288, r = i % 12288;
    int ch = r >> 11, t = (r >> 4) & 127, j = r & 15;
    int wv = t >> 5, lane = t & 31, half = lane >> 4;
    int ncol = wv * 16 + (lane & 15);
    int k = ch * 32 + frag_k(half, j >> 1, j & 1);
    int tap = k >> 6, c = k & 63;
    Wc[i] = (_Float16)convW[((l * HID + ncol) * HID + c) * 3 + tap];
  }
  // BN fold: y = conv*s + ((conv_b - m)*s + b)
  for (int i = tid0; i < NLAY * HID; i += str) {
    float s = bn_g[i] * rsqrtf(bn_v[i] + BN_EPS);
    scl[i] = s;
    sft[i] = (conv_b[i] - bn_m[i]) * s + bn_b[i];
  }
}

__global__ void k_deg(const int* __restrict__ ei, float* __restrict__ deg, int E, int N) {
  int e = blockIdx.x * blockDim.x + threadIdx.x;
  if (e >= E + N) return;
  int dst = (e < E) ? ei[E + e] : (e - E);
  atomic_add_f32(&deg[dst], 1.0f);
}

__global__ void k_dinv(float* deg, int N) {
  int i = blockIdx.x * blockDim.x + threadIdx.x;
  if (i < N) deg[i] = rsqrtf(deg[i]);
}

__global__ void k_coef(const int* __restrict__ ei, const float* __restrict__ dinv,
                       float* __restrict__ coef, int E, int N) {
  int e = blockIdx.x * blockDim.x + threadIdx.x;
  if (e >= E + N) return;
  int s = (e < E) ? ei[e] : (e - E);
  int d = (e < E) ? ei[E + e] : (e - E);
  coef[e] = dinv[s] * dinv[d];
}

// embedding: X0[n*768 + t*64 + h] = relu(be[h] + sum_c x[n,t,c]*We[c,h])
__global__ void k_embed(const float* __restrict__ x, const float* __restrict__ We,
                        const float* __restrict__ be, float* __restrict__ X0, int total) {
  int i = blockIdx.x * blockDim.x + threadIdx.x;
  int s = gridDim.x * blockDim.x;
  for (; i < total; i += s) {
    int h = i & 63;
    int nt = i >> 6;
    int t = nt % T_LEN, n = nt / T_LEN;
    float acc = be[h];
    const float* xr = x + n * (T_LEN * CIN_D) + t * CIN_D;
#pragma unroll
    for (int c = 0; c < CIN_D; ++c) acc += xr[c] * We[c * HID + h];
    X0[i] = fmaxf(acc, 0.0f);
  }
}

// init aggregate buffer with gcn bias: G[n*768 + d*12 + t] = gb[d]
__global__ void k_agg_init(float* __restrict__ G, const float* __restrict__ gb, int total) {
  int i = blockIdx.x * blockDim.x + threadIdx.x;
  int s = gridDim.x * blockDim.x;
  for (; i < total; i += s) {
    int d = (i % ROWF) / T_LEN;
    G[i] = gb[d];
  }
}

// scatter-add: one WAVE per (edge|self-loop), 8 edges / 256-thread block.
// All operands are L2-resident (61.4 MB src + 61.4 MB dst < 192 MB L2).
__global__ __launch_bounds__(256) void k_scatter(const int* __restrict__ ei,
                                                 const float* __restrict__ coef,
                                                 const float* __restrict__ Hs,
                                                 float* __restrict__ G, int E, int N) {
  int wv = threadIdx.x >> 5, lane = threadIdx.x & 31;
  int e = blockIdx.x * 8 + wv;
  if (e >= E + N) return;
  int s = (e < E) ? ei[e] : (e - E);
  int d = (e < E) ? ei[E + e] : (e - E);
  float co = coef[e];
  const float* hp = Hs + (size_t)s * ROWF;
  float* gp = G + (size_t)d * ROWF;
#pragma unroll 4
  for (int j = lane; j < ROWF; j += 32) atomic_add_f32(&gp[j], co * hp[j]);
}

// ------------------------- WMMA GEMM kernels -------------------------------
// ISA 7.12.2 16-bit A layout: lane l holds row m=l&15; its 16 halves for a
// K=32 chunk are two contiguous 16B runs at k = kb+8*(l>>4) and k+16.
// B fragments are pre-swizzled to fragment-major order by k_prep.
// C/D: lane l, slot u -> row m = u + 8*(l>>4), col = l&15.

// GCN linear: H[n*768 + d*12 + t] = sum_c Xi[n*768 + c*12 + t] * W[c][d]
__global__ __launch_bounds__(128) void k_gcn(const float* __restrict__ Xi,
                                             const _Float16* __restrict__ Wf,
                                             float* __restrict__ Ho, int NT) {
  __shared__ _Float16 Ash[16][72];
  const int tid = threadIdx.x;
  const int row0 = blockIdx.x * 16;
  const int n0 = row0 / T_LEN, t0 = row0 - n0 * T_LEN;
  const bool full = (row0 + 16) <= NT;  // uniform: hot path is branch-free
  if (full) {
    for (int i = tid; i < 16 * HID; i += 128) {
      int r = i >> 6, c = i & 63;
      int tt = t0 + r;
      int w = (tt >= T_LEN) + (tt >= 2 * T_LEN);
      Ash[r][c] = (_Float16)Xi[(size_t)(n0 + w) * ROWF + c * T_LEN + (tt - T_LEN * w)];
    }
  } else {
    for (int i = tid; i < 16 * HID; i += 128) {
      int r = i >> 6, c = i & 63;
      int tt = t0 + r;
      int w = (tt >= T_LEN) + (tt >= 2 * T_LEN);
      float v = 0.0f;
      if ((row0 + r) < NT) v = Xi[(size_t)(n0 + w) * ROWF + c * T_LEN + (tt - T_LEN * w)];
      Ash[r][c] = (_Float16)v;
    }
  }
  __syncthreads();
  const int lane = tid & 31, half = lane >> 4, m = lane & 15;
  const int ncol = ((tid >> 5) << 4) + (lane & 15);
  v8f acc = {};
#pragma unroll
  for (int ch = 0; ch < 2; ++ch) {
    const int kb = ch * 32;
    v8h alo = *(const v8h*)&Ash[m][kb + (half << 3)];
    v8h ahi = *(const v8h*)&Ash[m][kb + 16 + (half << 3)];
    v16h a = cat16(alo, ahi);
    v16h b = *(const v16h*)(Wf + ch * 2048 + tid * 16);
    acc = __builtin_amdgcn_wmma_f32_16x16x32_f16(false, a, false, b, (short)0, acc,
                                                 false, false);
  }
  if (full) {
#pragma unroll
    for (int u = 0; u < 8; ++u) {
      int tt = t0 + u + (half << 3);
      int w = (tt >= T_LEN) + (tt >= 2 * T_LEN);
      Ho[(size_t)(n0 + w) * ROWF + ncol * T_LEN + (tt - T_LEN * w)] = acc[u];
    }
  } else {
#pragma unroll
    for (int u = 0; u < 8; ++u) {
      int mm = u + (half << 3);
      int tt = t0 + mm;
      int w = (tt >= T_LEN) + (tt >= 2 * T_LEN);
      if ((row0 + mm) < NT)
        Ho[(size_t)(n0 + w) * ROWF + ncol * T_LEN + (tt - T_LEN * w)] = acc[u];
    }
  }
}

// temporal conv (K=192) + BN-fold + residual + relu
__global__ __launch_bounds__(128) void k_conv(const float* __restrict__ G,
                                              const float* __restrict__ Xin,
                                              const _Float16* __restrict__ Wf,
                                              const float* __restrict__ scl,
                                              const float* __restrict__ sft,
                                              float* __restrict__ Xout, int NT) {
  __shared__ _Float16 Ash[16][200];
  const int tid = threadIdx.x;
  const int row0 = blockIdx.x * 16;
  const int n0 = row0 / T_LEN, t0 = row0 - n0 * T_LEN;
  const bool full = (row0 + 16) <= NT;
  if (full) {
    for (int i = tid; i < 16 * KCONV; i += 128) {
      int r = i / KCONV, k = i - r * KCONV;
      int tap = k >> 6, c = k & 63;
      int tt = t0 + r;
      int w = (tt >= T_LEN) + (tt >= 2 * T_LEN);
      int tc = (tt - T_LEN * w) + tap - 1;  // temporal pad
      float v = 0.0f;
      if ((unsigned)tc < (unsigned)T_LEN)
        v = G[(size_t)(n0 + w) * ROWF + c * T_LEN + tc];
      Ash[r][k] = (_Float16)v;
    }
  } else {
    for (int i = tid; i < 16 * KCONV; i += 128) {
      int r = i / KCONV, k = i - r * KCONV;
      int tap = k >> 6, c = k & 63;
      int tt = t0 + r;
      int w = (tt >= T_LEN) + (tt >= 2 * T_LEN);
      int tc = (tt - T_LEN * w) + tap - 1;
      float v = 0.0f;
      if (((row0 + r) < NT) && (unsigned)tc < (unsigned)T_LEN)
        v = G[(size_t)(n0 + w) * ROWF + c * T_LEN + tc];
      Ash[r][k] = (_Float16)v;
    }
  }
  __syncthreads();
  const int lane = tid & 31, half = lane >> 4, m = lane & 15;
  const int ncol = ((tid >> 5) << 4) + (lane & 15);
  v8f acc = {};
#pragma unroll
  for (int ch = 0; ch < 6; ++ch) {
    const int kb = ch * 32;
    v8h alo = *(const v8h*)&Ash[m][kb + (half << 3)];
    v8h ahi = *(const v8h*)&Ash[m][kb + 16 + (half << 3)];
    v16h a = cat16(alo, ahi);
    v16h b = *(const v16h*)(Wf + ch * 2048 + tid * 16);
    acc = __builtin_amdgcn_wmma_f32_16x16x32_f16(false, a, false, b, (short)0, acc,
                                                 false, false);
  }
  const float s = scl[ncol], f = sft[ncol];
  if (full) {
#pragma unroll
    for (int u = 0; u < 8; ++u) {
      int tt = t0 + u + (half << 3);
      int w = (tt >= T_LEN) + (tt >= 2 * T_LEN);
      size_t idx = (size_t)(n0 + w) * ROWF + ncol * T_LEN + (tt - T_LEN * w);
      float y = acc[u] * s + f + Xin[idx];
      Xout[idx] = fmaxf(y, 0.0f);
    }
  } else {
#pragma unroll
    for (int u = 0; u < 8; ++u) {
      int mm = u + (half << 3);
      int tt = t0 + mm;
      int w = (tt >= T_LEN) + (tt >= 2 * T_LEN);
      if ((row0 + mm) < NT) {
        size_t idx = (size_t)(n0 + w) * ROWF + ncol * T_LEN + (tt - T_LEN * w);
        float y = acc[u] * s + f + Xin[idx];
        Xout[idx] = fmaxf(y, 0.0f);
      }
    }
  }
}

// ---------------- attention pooling + output MLP (1 wave / node) -----------

__global__ __launch_bounds__(256) void k_attn(
    const float* __restrict__ X,
    const float* __restrict__ ta_w1, const float* __restrict__ ta_b1,
    const float* __restrict__ ta_w2, const float* __restrict__ ta_b2,
    const float* __restrict__ fa_w1, const float* __restrict__ fa_b1,
    const float* __restrict__ fa_w2, const float* __restrict__ fa_b2,
    const float* __restrict__ ow1, const float* __restrict__ ob1,
    const float* __restrict__ ow2, const float* __restrict__ ob2,
    float* __restrict__ out, int Nn) {
  __shared__ float sW1[64 * 32], sO1[64 * 32];
  __shared__ float sTb1[32], sTa2[32], sOb1[32], sO2[32];
  __shared__ float sF1[12 * 6], sFb1[6], sF2[6];
  __shared__ float sScal[3];
  __shared__ float sxa[8][12 * 64];
  __shared__ float sxf[8][64];
  int tid = threadIdx.x;
  for (int i = tid; i < 2048; i += 256) { sW1[i] = ta_w1[i]; sO1[i] = ow1[i]; }
  if (tid < 32) { sTb1[tid] = ta_b1[tid]; sTa2[tid] = ta_w2[tid];
                  sOb1[tid] = ob1[tid];   sO2[tid]  = ow2[tid]; }
  if (tid < 72) sF1[tid] = fa_w1[tid];
  if (tid < 6)  { sFb1[tid] = fa_b1[tid]; sF2[tid] = fa_w2[tid]; }
  if (tid == 0) { sScal[0] = ta_b2[0]; sScal[1] = fa_b2[0]; sScal[2] = ob2[0]; }

  int wv = tid >> 5, lane = tid & 31;
  int node = blockIdx.x * 8 + wv;
  bool act = node < Nn;
  if (act) {
    const float* xp = X + (size_t)node * ROWF;
    for (int i = lane; i < ROWF; i += 32) sxa[wv][i] = xp[i];  // [t][h] view
  }
  __syncthreads();
  const float* xa = sxa[wv];

  // temporal attention weights (softmax over T)
  float tw[T_LEN];
#pragma unroll
  for (int t = 0; t < T_LEN; ++t) {
    float acc = sTb1[lane];
    const float* xt = xa + t * HID;
#pragma unroll
    for (int h = 0; h < HID; ++h) acc += xt[h] * sW1[h * 32 + lane];
    acc = fmaxf(acc, 0.0f) * sTa2[lane];
    tw[t] = wred_sum(acc) + sScal[0];
  }
  float mx = tw[0];
#pragma unroll
  for (int t = 1; t < T_LEN; ++t) mx = fmaxf(mx, tw[t]);
  float sm = 0.0f;
#pragma unroll
  for (int t = 0; t < T_LEN; ++t) { tw[t] = expf(tw[t] - mx); sm += tw[t]; }
  float ism = 1.0f / sm;
#pragma unroll
  for (int t = 0; t < T_LEN; ++t) tw[t] *= ism;

  // feature attention (softmax over H=64); lane owns h=lane and h=lane+32
  float fwv[2];
#pragma unroll
  for (int q = 0; q < 2; ++q) {
    int h = lane + q * 32;
    float a2 = sScal[1];
#pragma unroll
    for (int p = 0; p < 6; ++p) {
      float acc = sFb1[p];
#pragma unroll
      for (int t = 0; t < T_LEN; ++t) acc += xa[t * HID + h] * tw[t] * sF1[t * 6 + p];
      a2 += fmaxf(acc, 0.0f) * sF2[p];
    }
    fwv[q] = a2;
  }
  float m2 = wred_max(fmaxf(fwv[0], fwv[1]));
  float e0 = expf(fwv[0] - m2), e1 = expf(fwv[1] - m2);
  float inv2 = 1.0f / wred_sum(e0 + e1);

  // pooled features
#pragma unroll
  for (int q = 0; q < 2; ++q) {
    int h = lane + q * 32;
    float xf = 0.0f;
#pragma unroll
    for (int t = 0; t < T_LEN; ++t) xf += xa[t * HID + h] * tw[t];
    xf *= (q == 0 ? e0 : e1) * inv2;
    if (act) sxf[wv][h] = xf;
  }
  __syncthreads();

  // output MLP
  float acc = sOb1[lane];
#pragma unroll
  for (int h = 0; h < HID; ++h) acc += sxf[wv][h] * sO1[h * 32 + lane];
  acc = fmaxf(acc, 0.0f) * sO2[lane];
  float res = wred_sum(acc) + sScal[2];
  if (act && lane == 0) out[node] = res;
}

// ------------------------------- launcher ----------------------------------

extern "C" void kernel_launch(void* const* d_in, const int* in_sizes, int n_in,
                              void* d_out, int out_size, void* d_ws, size_t ws_size,
                              hipStream_t stream) {
  const float* x       = (const float*)d_in[0];
  const int*   ei      = (const int*)d_in[1];
  const float* We      = (const float*)d_in[2];
  const float* be      = (const float*)d_in[3];
  const float* gcn_W   = (const float*)d_in[4];
  const float* gcn_b   = (const float*)d_in[5];
  const float* conv_w  = (const float*)d_in[6];
  const float* conv_b  = (const float*)d_in[7];
  const float* bn_g    = (const float*)d_in[8];
  const float* bn_b    = (const float*)d_in[9];
  const float* bn_m    = (const float*)d_in[10];
  const float* bn_v    = (const float*)d_in[11];
  const float* ta_w1   = (const float*)d_in[12];
  const float* ta_b1   = (const float*)d_in[13];
  const float* ta_w2   = (const float*)d_in[14];
  const float* ta_b2   = (const float*)d_in[15];
  const float* fa_w1   = (const float*)d_in[16];
  const float* fa_b1   = (const float*)d_in[17];
  const float* fa_w2   = (const float*)d_in[18];
  const float* fa_b2   = (const float*)d_in[19];
  const float* ow1     = (const float*)d_in[20];
  const float* ob1     = (const float*)d_in[21];
  const float* ow2     = (const float*)d_in[22];
  const float* ob2     = (const float*)d_in[23];

  const int N  = in_sizes[0] / (T_LEN * CIN_D);  // 20000
  const int E  = in_sizes[1] / 2;                // 320000
  const int NT = N * T_LEN;                      // 240000
  const int EN = E + N;
  const size_t NTH = (size_t)N * ROWF;

  // ---- carve workspace (256B aligned) ----
  char* base = (char*)d_ws;
  size_t off = 0;
  auto carve = [&](size_t bytes) -> void* {
    off = (off + 255) & ~(size_t)255;
    void* p = base + off;
    off += bytes;
    return p;
  };
  float*    deg  = (float*)carve((size_t)N * 4);
  float*    coef = (float*)carve((size_t)EN * 4);
  _Float16* Wg   = (_Float16*)carve((size_t)NLAY * 2 * 128 * 16 * 2);
  _Float16* Wc   = (_Float16*)carve((size_t)NLAY * 6 * 128 * 16 * 2);
  float*    scl  = (float*)carve((size_t)NLAY * HID * 4);
  float*    sft  = (float*)carve((size_t)NLAY * HID * 4);
  float*    X0   = (float*)carve(NTH * 4);
  float*    B1   = (float*)carve(NTH * 4);
  float*    Gb   = (float*)carve(NTH * 4);
  (void)ws_size; (void)n_in; (void)out_size;

  // ---- weight prep + graph normalization ----
  k_prep<<<96, 256, 0, stream>>>(gcn_W, conv_w, bn_g, bn_b, bn_m, bn_v, conv_b,
                                 Wg, Wc, scl, sft);
  k_fill<<<256, 256, 0, stream>>>(deg, N, 0.0f);
  k_deg<<<(EN + 255) / 256, 256, 0, stream>>>(ei, deg, E, N);
  k_dinv<<<(N + 255) / 256, 256, 0, stream>>>(ei ? deg : deg, N);
  k_coef<<<(EN + 255) / 256, 256, 0, stream>>>(ei, deg, coef, E, N);

  // ---- embedding ----
  k_embed<<<4096, 256, 0, stream>>>(x, We, be, X0, NT * HID);

  // ---- ST layers ----
  const int tiles = (NT + 15) / 16;
  float* bufs[2] = {X0, B1};
  for (int l = 0; l < NLAY; ++l) {
    float* Xi = bufs[l & 1];
    float* Xo = bufs[(l + 1) & 1];  // also holds H (dead before conv writes)
    k_gcn<<<tiles, 128, 0, stream>>>(Xi, Wg + (size_t)l * 4096, Xo, NT);
    k_agg_init<<<4096, 256, 0, stream>>>(Gb, gcn_b + l * HID, NT * HID);
    k_scatter<<<(EN + 7) / 8, 256, 0, stream>>>(ei, coef, Xo, Gb, E, N);
    k_conv<<<tiles, 128, 0, stream>>>(Gb, Xi, Wc + (size_t)l * 12288,
                                      scl + l * HID, sft + l * HID, Xo, NT);
  }
  float* Xfin = bufs[NLAY & 1];  // layer 2 output -> B1

  // ---- attention pooling + output MLP ----
  k_attn<<<(N + 7) / 8, 256, 0, stream>>>(Xfin, ta_w1, ta_b1, ta_w2, ta_b2,
                                          fa_w1, fa_b1, fa_w2, fa_b2,
                                          ow1, ob1, ow2, ob2,
                                          (float*)d_out, N);
}
